// TalaThreadingLayer_70463233458679
// MI455X (gfx1250) — matrix operand order repaired and checked by reading, not compiled, not run on hardware
//
#include <hip/hip_runtime.h>
#include <hip/hip_bf16.h>
#include <math.h>

// ---------------------------------------------------------------------------
// TalaThreadingLayer for MI455X (gfx1250, wave32, WMMA + TDM)
// B=4, T=2048, D=512, K(threads)=4, CYCLES = {5,6,7,8} = k+5
// ---------------------------------------------------------------------------

typedef __attribute__((ext_vector_type(16))) _Float16 v16h;
typedef __attribute__((ext_vector_type(8)))  _Float16 v8h;
typedef __attribute__((ext_vector_type(8)))  float    v8f;
typedef __attribute__((ext_vector_type(4)))  unsigned int u32x4;
typedef __attribute__((ext_vector_type(8)))  int i32x8;
typedef __attribute__((ext_vector_type(4)))  int i32x4;

#define BB 4
#define TT 2048
#define DD 512
#define D3 1536
#define BT 8192            // B*T

__device__ __forceinline__ v16h mk16(v8h lo, v8h hi) {
  return __builtin_shufflevector(lo, hi, 0,1,2,3,4,5,6,7,8,9,10,11,12,13,14,15);
}

// D(f32 16x16) = A(f16 16x32) * B(f16 32x16) + C
__device__ __forceinline__ v8f wmma16(v16h a, v16h b, v8f c) {
  return __builtin_amdgcn_wmma_f32_16x16x32_f16(
      /*neg_a=*/false, a, /*neg_b=*/false, b,
      /*c_mod=*/(short)0, c, /*reuse_a=*/false, /*reuse_b=*/false);
}

// ---------------------------------------------------------------------------
// Tensor Data Mover: DMA a 128x32 f16 tile (row stride = rowStrideElems in
// global) into LDS at lds_byte_off, padding each 64B row by 32B so the LDS
// row stride is 96B (= LSTR f16 elems), matching the fragment loaders.
// D# bitfields per CDNA5 ISA.
// ---------------------------------------------------------------------------
#define HAVE_TDM __has_builtin(__builtin_amdgcn_tensor_load_to_lds)

#if HAVE_TDM
__device__ __forceinline__ void tdm_load_tile_128x32_f16(
    const _Float16* gsrc, unsigned lds_byte_off, int rowStrideElems) {
  unsigned long long ga = (unsigned long long)gsrc;
  u32x4 g0;
  g0[0] = 1u;                                   // count=1 (valid user D#)
  g0[1] = lds_byte_off;                         // lds_addr (bytes)
  g0[2] = (unsigned)(ga & 0xffffffffu);         // global_addr[31:0]
  g0[3] = (unsigned)((ga >> 32) & 0x1ffffffu)   // global_addr[56:32]
          | (2u << 30);                         // type = 2 ("image")
  i32x8 g1;
  g1[0] = (1 << 16)        // data_size = 2 bytes
        | (1 << 20)        // pad_enable
        | (3 << 22)        // pad_interval: 16 DWORDs (one 64B row)
        | (7 << 25);       // pad_amount:   8 DWORDs (32B) -> 96B LDS stride
  g1[1] = (int)(32u << 16);        // tensor_dim0 = 32 (lo16 in dw1[31:16])
  g1[2] = (int)(128u << 16);       // tensor_dim0 hi = 0 ; tensor_dim1 = 128
  g1[3] = (int)(32u << 16);        // tensor_dim1 hi = 0 ; tile_dim0 = 32
  g1[4] = 128;                     // tile_dim1 = 128 ; tile_dim2 = 0 (unused)
  g1[5] = rowStrideElems;          // tensor_dim0_stride[31:0] (elements)
  g1[6] = 0;                       // stride hi ; tensor_dim1_stride lo
  g1[7] = 0;
  i32x4 z4 = {0, 0, 0, 0};         // groups 2/3: dims 2..4 unused
#if __clang_major__ >= 23
  i32x8 z8 = {0, 0, 0, 0, 0, 0, 0, 0};
  __builtin_amdgcn_tensor_load_to_lds(g0, g1, z4, z4, z8, 0);
#else
  __builtin_amdgcn_tensor_load_to_lds(g0, g1, z4, z4, 0);
#endif
}
#endif

// ---------------------------------------------------------------------------
// Stage 0: x_pe = x + 0.5*pe + 0.5*tala ; emit fp32 and f16 copies
// ---------------------------------------------------------------------------
__global__ __launch_bounds__(256)
void pe_add_kernel(const float* __restrict__ x, float* __restrict__ xpe,
                   _Float16* __restrict__ xpeh) {
  int idx = blockIdx.x * 256 + threadIdx.x;
  if (idx >= BB * TT * DD) return;
  int d = idx & (DD - 1);
  int t = (idx >> 9) & (TT - 1);
  float pos = (float)t;
  int i = d >> 1;
  float dv = expf((float)(2 * i) * (-0.017988946039015984f));  // -ln(10000)/512
  float pe = (d & 1) ? cosf(pos * dv) : sinf(pos * dv);
  int sec = d >> 7, off = d & 127;
  int ci = sec + 5;
  float phase = ((float)(t % ci)) * (6.283185307179586f / (float)ci);
  float harm = (float)((off & 63) + 1);
  float tala = (off < 64) ? sinf(phase * harm) : cosf(phase * harm);
  float val = x[idx] + 0.5f * (pe + tala);
  xpe[idx] = val;
  xpeh[idx] = (_Float16)val;
}

__global__ __launch_bounds__(256)
void cvt_f32_f16_kernel(const float* __restrict__ in, _Float16* __restrict__ out, int n) {
  int i = blockIdx.x * 256 + threadIdx.x;
  if (i < n) out[i] = (_Float16)in[i];
}

// ---------------------------------------------------------------------------
// Generic WMMA GEMM:  C[m,n] = sum_k A[m,k] * W[n,k]  (+ bias[n])
// A: (M,K) f16 row-major, W: (N,K) f16 row-major. 128x128 tile / workgroup,
// 8 waves each computing a 32x64 sub-tile. K staged in steps of 32 through
// DOUBLE-BUFFERED LDS via the Tensor Data Mover: one barrier per K-step,
// wave 0 issues the next tile's DMA while all waves run WMMA on the current
// tile; TENSORcnt hides the L2->LDS transfer behind compute.
// ---------------------------------------------------------------------------
#define LSTR 48  // padded LDS row stride (f16 elems): 96B, 16B-aligned chunks

__global__ __launch_bounds__(256)
void gemm_f16_wmma(const _Float16* __restrict__ A, long long aBatch,
                   const _Float16* __restrict__ W, long long wBatch,
                   const float* __restrict__ bias, long long biasBatch,
                   void* __restrict__ Cout, long long cBatch, int cIsF16,
                   int M, int N, int K) {
  __shared__ _Float16 lA[2][128 * LSTR];
  __shared__ _Float16 lW[2][128 * LSTR];

  const int bz = blockIdx.z;
  const _Float16* Ab = A + (size_t)aBatch * bz;
  const _Float16* Wb = W + (size_t)wBatch * bz;
  const float* biasb = bias ? (bias + (size_t)biasBatch * bz) : nullptr;
  const int m0 = blockIdx.y * 128, n0 = blockIdx.x * 128;

  const int tid = threadIdx.x;
  const int wid = tid >> 5, lane = tid & 31;
  const int half = lane >> 4, lr = lane & 15;
  const int wm = (wid & 3) * 32;   // wave M offset in tile
  const int wn = (wid >> 2) * 64;  // wave N offset in tile

  v8f zero = {};
  v8f acc[2][4];
  for (int i = 0; i < 2; ++i)
    for (int j = 0; j < 4; ++j) acc[i][j] = zero;

#if HAVE_TDM
  const unsigned ldsA0 = (unsigned)(size_t)(&lA[0][0]);  // LDS byte offsets
  const unsigned ldsA1 = (unsigned)(size_t)(&lA[1][0]);
  const unsigned ldsW0 = (unsigned)(size_t)(&lW[0][0]);
  const unsigned ldsW1 = (unsigned)(size_t)(&lW[1][0]);
#endif

  auto stage = [&](int kk, int p) {
#if HAVE_TDM
    if (tid < 32) {  // wave 0 drives the DMA (async; waited next iteration)
      tdm_load_tile_128x32_f16(Ab + (size_t)m0 * K + kk, p ? ldsA1 : ldsA0, K);
      tdm_load_tile_128x32_f16(Wb + (size_t)n0 * K + kk, p ? ldsW1 : ldsW0, K);
    }
#else
    int row0 = tid >> 2, col = (tid & 3) << 3;
    int row1 = row0 + 64;
    uint4 ra0 = *(const uint4*)&Ab[(size_t)(m0 + row0) * K + kk + col];
    uint4 rw0 = *(const uint4*)&Wb[(size_t)(n0 + row0) * K + kk + col];
    uint4 ra1 = *(const uint4*)&Ab[(size_t)(m0 + row1) * K + kk + col];
    uint4 rw1 = *(const uint4*)&Wb[(size_t)(n0 + row1) * K + kk + col];
    *(uint4*)&lA[p][row0 * LSTR + col] = ra0;
    *(uint4*)&lW[p][row0 * LSTR + col] = rw0;
    *(uint4*)&lA[p][row1 * LSTR + col] = ra1;
    *(uint4*)&lW[p][row1 * LSTR + col] = rw1;
#endif
  };

  stage(0, 0);  // preload first tile into buffer 0
  int p = 0;
  for (int kk = 0; kk < K; kk += 32) {
#if HAVE_TDM
    if (tid < 32) __builtin_amdgcn_s_wait_tensorcnt(0);  // tile p resident
#endif
    // Single barrier per K-step: publishes buffer p to all waves AND
    // guarantees everyone finished reading buffer p^1 (previous step).
    __syncthreads();
    if (kk + 32 < K) stage(kk + 32, p ^ 1);  // overlap DMA with WMMA below

    // Build fragments. A (16x32 f16): lane lr row m, half h: K = j+8h (j<8),
    // j+8+8h (j>=8). B (32x16): lane lr col n, half h: same K order as A.
    v16h af[2], bf[4];
#pragma unroll
    for (int i = 0; i < 2; ++i) {
      const _Float16* r = &lA[p][(wm + i * 16 + lr) * LSTR];
      af[i] = mk16(*(const v8h*)(r + 8 * half), *(const v8h*)(r + 16 + 8 * half));
    }
#pragma unroll
    for (int j = 0; j < 4; ++j) {
      const _Float16* r = &lW[p][(wn + j * 16 + lr) * LSTR];
      bf[j] = mk16(*(const v8h*)(r + 8 * half), *(const v8h*)(r + 16 + 8 * half));
    }
#pragma unroll
    for (int i = 0; i < 2; ++i)
#pragma unroll
      for (int j = 0; j < 4; ++j) acc[i][j] = wmma16(af[i], bf[j], acc[i][j]);
    p ^= 1;
  }

  // Epilogue. D layout: VGPR r, lane l -> m = r + 8*(l/16), n = l%16
#pragma unroll
  for (int j = 0; j < 4; ++j) {
    int n = n0 + wn + j * 16 + lr;
    float bv = biasb ? biasb[n] : 0.f;
#pragma unroll
    for (int i = 0; i < 2; ++i) {
      int mbase = m0 + wm + i * 16 + 8 * half;
#pragma unroll
      for (int r = 0; r < 8; ++r) {
        size_t idx = (size_t)cBatch * bz + (size_t)(mbase + r) * N + n;
        float v = acc[i][j][r] + bv;
        if (cIsF16) ((_Float16*)Cout)[idx] = (_Float16)v;
        else        ((float*)Cout)[idx] = v;
      }
    }
  }
}

// ---------------------------------------------------------------------------
// GRU recurrence: one block per tala-thread k; h in LDS; per step compute
// gh(1536x4) = whh_k(1536x512) * h^T(512x4) via WMMA, then gate update.
// ---------------------------------------------------------------------------
#define HBS 520  // padded f16 h row stride (1040B, 16B multiple)

__global__ __launch_bounds__(256)
void gru_scan_kernel(const _Float16* __restrict__ whh_h,  // (4,1536,512) f16
                     const _Float16* __restrict__ xp,     // (4,BT,1536) f16
                     const float* __restrict__ bhh,       // (4,1536)
                     _Float16* __restrict__ comb) {       // (4,BT,1024), gru half
  __shared__ float    gh[D3 * 4];       // gh[i*4 + b]
  __shared__ float    hf[4 * DD];       // fp32 h
  __shared__ _Float16 hb[16 * HBS];     // f16 h as B operand (rows 4..15 zero)

  const int k = blockIdx.x;
  const int tid = threadIdx.x, wid = tid >> 5, lane = tid & 31;
  const int half = lane >> 4, lr = lane & 15;

  for (int i = tid; i < 4 * DD; i += 256) hf[i] = 0.f;
  for (int i = tid; i < 16 * HBS; i += 256) hb[i] = (_Float16)0.f;
  __syncthreads();

  const _Float16* wkb = whh_h + (size_t)k * D3 * DD;
  const float* bhhk = bhh + k * D3;
  const _Float16* xpk = xp + (size_t)k * BT * D3;
  _Float16* combk = comb + (size_t)k * BT * 1024;

#pragma unroll 1
  for (int t = 0; t < TT; ++t) {
    // ---- gh = whh * h^T (M=1536 rows, N cols 0..3 valid), 12 M-tiles/wave
#pragma unroll 1
    for (int mi = 0; mi < 12; ++mi) {
      int mt = wid * 12 + mi;
      const _Float16* arow = wkb + (size_t)(mt * 16 + lr) * DD;
      v8f acc = {};
#pragma unroll 4
      for (int ks = 0; ks < 16; ++ks) {
        v16h a = mk16(*(const v8h*)(arow + ks * 32 + 8 * half),
                      *(const v8h*)(arow + ks * 32 + 16 + 8 * half));
        const _Float16* hrow = &hb[lr * HBS + ks * 32];
        v16h b = mk16(*(const v8h*)(hrow + 8 * half), *(const v8h*)(hrow + 16 + 8 * half));
        acc = wmma16(a, b, acc);
      }
      if (lr < 4) {
#pragma unroll
        for (int r = 0; r < 8; ++r)
          gh[(mt * 16 + 8 * half + r) * 4 + lr] = acc[r];
      }
    }
    __syncthreads();

    // ---- gates + state update: 4*512 = 2048 elems
    for (int u = tid; u < 4 * DD; u += 256) {
      int b = u >> 9, d = u & (DD - 1);
      const _Float16* xr_ = xpk + (size_t)(b * TT + t) * D3;
      float xr = (float)xr_[d];
      float xz = (float)xr_[DD + d];
      float xn = (float)xr_[2 * DD + d];
      float hr = gh[d * 4 + b] + bhhk[d];
      float hz = gh[(DD + d) * 4 + b] + bhhk[DD + d];
      float hn = gh[(2 * DD + d) * 4 + b] + bhhk[2 * DD + d];
      float r = 1.f / (1.f + __expf(-(xr + hr)));
      float z = 1.f / (1.f + __expf(-(xz + hz)));
      float pre = xn + r * hn;
      float e = __expf(-2.f * pre);
      float n = (1.f - e) / (1.f + e);  // tanh(pre)
      float hp = hf[b * DD + d];
      float hnew = (1.f - z) * n + z * hp;
      hf[b * DD + d] = hnew;
      hb[b * HBS + d] = (_Float16)hnew;
      combk[(size_t)(b * TT + t) * 1024 + d] = (_Float16)hnew;
    }
    __syncthreads();
  }
}

// ---------------------------------------------------------------------------
// Banded attention: window of c=k+5 positions; one wave per (k,b,t)
// ---------------------------------------------------------------------------
__global__ __launch_bounds__(256)
void tala_attn_kernel(const _Float16* __restrict__ q, const _Float16* __restrict__ kmat,
                      const _Float16* __restrict__ v, _Float16* __restrict__ comb) {
  const int tid = threadIdx.x, wid = tid >> 5, lane = tid & 31;
  const int t = blockIdx.x * 8 + wid;
  const int kz = blockIdx.z, b = blockIdx.y;
  const int c = 5 + kz;
  const size_t base = ((size_t)kz * BT + (size_t)b * TT) * DD;

  float qv[16];
  {
    const _Float16* qrow = q + base + (size_t)t * DD + lane * 16;
    v8h q0 = *(const v8h*)qrow;
    v8h q1 = *(const v8h*)(qrow + 8);
#pragma unroll
    for (int j = 0; j < 8; ++j) { qv[j] = (float)q0[j]; qv[8 + j] = (float)q1[j]; }
  }

  float sc[8];
#pragma unroll
  for (int w = 0; w < 8; ++w) {
    bool valid = (w < c) && (t - w >= 0);   // wave-uniform
    float p = -1e30f;
    if (valid) {
      const _Float16* krow = kmat + base + (size_t)(t - w) * DD + lane * 16;
      v8h k0 = *(const v8h*)krow;
      v8h k1 = *(const v8h*)(krow + 8);
      p = 0.f;
#pragma unroll
      for (int j = 0; j < 8; ++j)
        p += qv[j] * (float)k0[j] + qv[8 + j] * (float)k1[j];
      for (int off = 16; off; off >>= 1) p += __shfl_xor(p, off);
      p *= 0.04419417382415922f;  // 1/sqrt(512)
    }
    sc[w] = p;
  }

  float mx = sc[0];
#pragma unroll
  for (int w = 1; w < 8; ++w) mx = fmaxf(mx, sc[w]);
  float wt[8], s = 0.f;
#pragma unroll
  for (int w = 0; w < 8; ++w) { wt[w] = __expf(sc[w] - mx); s += wt[w]; }
  float inv = 1.f / s;

  float acc[16] = {};
#pragma unroll
  for (int w = 0; w < 8; ++w) {
    if (sc[w] > -1e29f) {
      float g = wt[w] * inv;
      const _Float16* vrow = v + base + (size_t)(t - w) * DD + lane * 16;
      v8h v0 = *(const v8h*)vrow;
      v8h v1 = *(const v8h*)(vrow + 8);
#pragma unroll
      for (int j = 0; j < 8; ++j) {
        acc[j] += g * (float)v0[j];
        acc[8 + j] += g * (float)v1[j];
      }
    }
  }
  _Float16* orow = comb + ((size_t)kz * BT + (size_t)b * TT + t) * 1024 + DD + lane * 16;
#pragma unroll
  for (int j = 0; j < 16; ++j) orow[j] = (_Float16)acc[j];
}

// ---------------------------------------------------------------------------
// LayerNorm helpers / kernels (row length 512, block = 256 threads)
// ---------------------------------------------------------------------------
__device__ __forceinline__ float blockSum256(float v, float* sm) {
  for (int off = 16; off; off >>= 1) v += __shfl_xor(v, off);
  int wid = threadIdx.x >> 5;
  if ((threadIdx.x & 31) == 0) sm[wid] = v;
  __syncthreads();
  if (threadIdx.x == 0) {
    float s = 0.f;
    for (int i = 0; i < 8; ++i) s += sm[i];
    sm[8] = s;
  }
  __syncthreads();
  float r = sm[8];
  __syncthreads();
  return r;
}

// thr = LN(beat + x_pe; g_k, b_k) -> merged[bt, k*512 + d] (f16)
__global__ __launch_bounds__(256)
void ln_thr_kernel(const float* __restrict__ beat, const float* __restrict__ xpe,
                   const float* __restrict__ g, const float* __restrict__ bb,
                   _Float16* __restrict__ merged) {
  __shared__ float sm[16];
  const int bt = blockIdx.x, k = blockIdx.y, tid = threadIdx.x;
  const float* brow = beat + ((size_t)k * BT + bt) * DD;
  const float* xrow = xpe + (size_t)bt * DD;
  float v0 = brow[tid] + xrow[tid];
  float v1 = brow[tid + 256] + xrow[tid + 256];
  float mean = blockSum256(v0 + v1, sm) * (1.f / 512.f);
  float d0 = v0 - mean, d1 = v1 - mean;
  float var = blockSum256(d0 * d0 + d1 * d1, sm) * (1.f / 512.f);
  float rstd = rsqrtf(var + 1e-5f);
  float y0 = d0 * rstd * g[k * DD + tid] + bb[k * DD + tid];
  float y1 = d1 * rstd * g[k * DD + tid + 256] + bb[k * DD + tid + 256];
  merged[(size_t)bt * (4 * DD) + k * DD + tid] = (_Float16)y0;
  merged[(size_t)bt * (4 * DD) + k * DD + tid + 256] = (_Float16)y1;
}

// out = LN(mix + x_pe; bn_g, bn_b) -> d_out (fp32)
__global__ __launch_bounds__(256)
void ln_final_kernel(const float* __restrict__ mix, const float* __restrict__ xpe,
                     const float* __restrict__ g, const float* __restrict__ bb,
                     float* __restrict__ out) {
  __shared__ float sm[16];
  const int bt = blockIdx.x, tid = threadIdx.x;
  const float* mrow = mix + (size_t)bt * DD;
  const float* xrow = xpe + (size_t)bt * DD;
  float v0 = mrow[tid] + xrow[tid];
  float v1 = mrow[tid + 256] + xrow[tid + 256];
  float mean = blockSum256(v0 + v1, sm) * (1.f / 512.f);
  float d0 = v0 - mean, d1 = v1 - mean;
  float var = blockSum256(d0 * d0 + d1 * d1, sm) * (1.f / 512.f);
  float rstd = rsqrtf(var + 1e-5f);
  out[(size_t)bt * DD + tid] = d0 * rstd * g[tid] + bb[tid];
  out[(size_t)bt * DD + tid + 256] = d1 * rstd * g[tid + 256] + bb[tid + 256];
}

// ---------------------------------------------------------------------------
// Host orchestration
// ---------------------------------------------------------------------------
extern "C" void kernel_launch(void* const* d_in, const int* in_sizes, int n_in,
                              void* d_out, int out_size, void* d_ws, size_t ws_size,
                              hipStream_t stream) {
  const float* x     = (const float*)d_in[0];
  const float* wih   = (const float*)d_in[1];
  const float* whh   = (const float*)d_in[2];
  const float* bih   = (const float*)d_in[3];
  const float* bhh   = (const float*)d_in[4];
  const float* wq    = (const float*)d_in[5];
  const float* wk    = (const float*)d_in[6];
  const float* wv    = (const float*)d_in[7];
  const float* wbeat = (const float*)d_in[8];
  const float* ng    = (const float*)d_in[9];
  const float* nb    = (const float*)d_in[10];
  const float* wmix  = (const float*)d_in[11];
  const float* bng   = (const float*)d_in[12];
  const float* bnb   = (const float*)d_in[13];

  char* ws = (char*)d_ws;
  size_t off = 0;
  auto alloc = [&](size_t bytes) -> void* {
    void* p = ws + off;
    off += (bytes + 255) & ~(size_t)255;
    return p;
  };

  float*    xpe    = (float*)alloc((size_t)BT * DD * 4);
  _Float16* xpeh   = (_Float16*)alloc((size_t)BT * DD * 2);
  _Float16* wih_h  = (_Float16*)alloc((size_t)4 * D3 * DD * 2);
  _Float16* whh_h  = (_Float16*)alloc((size_t)4 * D3 * DD * 2);
  _Float16* wq_h   = (_Float16*)alloc((size_t)4 * DD * DD * 2);
  _Float16* wk_h   = (_Float16*)alloc((size_t)4 * DD * DD * 2);
  _Float16* wv_h   = (_Float16*)alloc((size_t)4 * DD * DD * 2);
  _Float16* wbt_h  = (_Float16*)alloc((size_t)4 * DD * 2 * DD * 2);
  _Float16* wmx_h  = (_Float16*)alloc((size_t)DD * 4 * DD * 2);
  _Float16* xp_h   = (_Float16*)alloc((size_t)4 * BT * D3 * 2);  // later reused as beat f32
  _Float16* qh     = (_Float16*)alloc((size_t)4 * BT * DD * 2);  // later reused as merged f16
  _Float16* kh     = (_Float16*)alloc((size_t)4 * BT * DD * 2);  // later reused as mix f32
  _Float16* vh     = (_Float16*)alloc((size_t)4 * BT * DD * 2);
  _Float16* comb   = (_Float16*)alloc((size_t)4 * BT * 2 * DD * 2);

  // dead-buffer reuse (strictly after last read of the aliased region)
  float*    beat   = (float*)xp_h;    // 4*BT*512*4 = 64MB <= 96MB
  _Float16* merged = qh;              // BT*2048*2  = 32MB == 32MB
  float*    mixo   = (float*)kh;      // BT*512*4   = 16MB <= 32MB

  // 0) positional encodings (+fp16 copy of x_pe)
  pe_add_kernel<<<(BB * TT * DD) / 256, 256, 0, stream>>>(x, xpe, xpeh);

  // 1) weight conversions fp32 -> fp16
  auto cvt = [&](const float* src, _Float16* dst, int n) {
    cvt_f32_f16_kernel<<<(n + 255) / 256, 256, 0, stream>>>(src, dst, n);
  };
  cvt(wih,   wih_h, 4 * D3 * DD);
  cvt(whh,   whh_h, 4 * D3 * DD);
  cvt(wq,    wq_h,  4 * DD * DD);
  cvt(wk,    wk_h,  4 * DD * DD);
  cvt(wv,    wv_h,  4 * DD * DD);
  cvt(wbeat, wbt_h, 4 * DD * 2 * DD);
  cvt(wmix,  wmx_h, DD * 4 * DD);

  // 2) GRU input projection: xp[k] = x_pe * wih_k^T + bih_k  (f16 out)
  gemm_f16_wmma<<<dim3(D3 / 128, BT / 128, 4), 256, 0, stream>>>(
      xpeh, 0LL, wih_h, (long long)D3 * DD, bih, (long long)D3,
      xp_h, (long long)BT * D3, 1, BT, D3, DD);

  // 3) q/k/v projections (f16 out)
  gemm_f16_wmma<<<dim3(DD / 128, BT / 128, 4), 256, 0, stream>>>(
      xpeh, 0LL, wq_h, (long long)DD * DD, nullptr, 0LL,
      qh, (long long)BT * DD, 1, BT, DD, DD);
  gemm_f16_wmma<<<dim3(DD / 128, BT / 128, 4), 256, 0, stream>>>(
      xpeh, 0LL, wk_h, (long long)DD * DD, nullptr, 0LL,
      kh, (long long)BT * DD, 1, BT, DD, DD);
  gemm_f16_wmma<<<dim3(DD / 128, BT / 128, 4), 256, 0, stream>>>(
      xpeh, 0LL, wv_h, (long long)DD * DD, nullptr, 0LL,
      vh, (long long)BT * DD, 1, BT, DD, DD);

  // 4) sequential GRU scan (writes gru half of comb)
  gru_scan_kernel<<<4, 256, 0, stream>>>(whh_h, xp_h, bhh, comb);

  // 5) banded attention (writes attn half of comb)
  tala_attn_kernel<<<dim3(TT / 8, BB, 4), 256, 0, stream>>>(qh, kh, vh, comb);

  // 6) beat[k] = comb[k] * w_beat_k^T  (fp32 out; reuses xp region)
  gemm_f16_wmma<<<dim3(DD / 128, BT / 128, 4), 256, 0, stream>>>(
      comb, (long long)BT * 2 * DD, wbt_h, (long long)DD * 2 * DD, nullptr, 0LL,
      beat, (long long)BT * DD, 0, BT, DD, 2 * DD);

  // 7) thr = LN(beat + x_pe) -> merged (B,T,4D) f16
  ln_thr_kernel<<<dim3(BT, 4), 256, 0, stream>>>(beat, xpe, ng, nb, merged);

  // 8) mix = merged * w_mix^T  (fp32 out)
  gemm_f16_wmma<<<dim3(DD / 128, BT / 128, 1), 256, 0, stream>>>(
      merged, 0LL, wmx_h, 0LL, nullptr, 0LL,
      mixo, 0LL, 0, BT, DD, 4 * DD);

  // 9) out = LN(mix + x_pe)
  ln_final_kernel<<<BT, 256, 0, stream>>>(mixo, xpe, bng, bnb, (float*)d_out);
}
